// CCN1D_76321568850347
// MI455X (gfx1250) — matrix operand order repaired and checked by reading, not compiled, not run on hardware
//
#include <hip/hip_runtime.h>
#include <stdint.h>

typedef __attribute__((ext_vector_type(16))) _Float16     v16h;
typedef __attribute__((ext_vector_type(8)))  float        v8f;
typedef __attribute__((ext_vector_type(4)))  unsigned int v4u;
typedef __attribute__((ext_vector_type(4)))  unsigned int u32x4;
typedef __attribute__((ext_vector_type(8)))  int          i32x8;
typedef __attribute__((ext_vector_type(4)))  int          i32x4;

#define NV    10000
#define NPAD  10112     // 79 * 128 rows
#define C_IN  512
#define MSG   64
#define NCLS  16
#define TOTW  384       // dense(128) + s0f(64) + s1f(64) + s0r(64) + s1r(64)

#ifndef __has_builtin
#define __has_builtin(x) 0
#endif
#if __has_builtin(__builtin_amdgcn_tensor_load_to_lds) && \
    __has_builtin(__builtin_amdgcn_s_wait_tensorcnt)
#define HAVE_TDM 1
#else
#define HAVE_TDM 0
#endif

union FragH { v16h v; v4u q[2]; };

// ---------- f32 -> f16 convert with optional zero row padding ----------
__global__ void cvt_f32_f16_pad(const float* __restrict__ src, _Float16* __restrict__ dst,
                                int rows_valid, int rows_total, int cols) {
  int idx = blockIdx.x * blockDim.x + threadIdx.x;
  if (idx >= rows_total * cols) return;
  int r = idx / cols;
  float v = (r < rows_valid) ? src[idx] : 0.0f;
  dst[idx] = (_Float16)v;
}

// ---------- w0a_eff = (w0a[:128] + w0a[128:]) / 13  (fold concat([g,g])/13) ----------
__global__ void make_w0a_eff(const float* __restrict__ w, _Float16* __restrict__ out) {
  int idx = blockIdx.x * blockDim.x + threadIdx.x;
  if (idx >= 128 * 128) return;
  int k = idx >> 7, j = idx & 127;
  out[idx] = (_Float16)((w[k * 128 + j] + w[(k + 128) * 128 + j]) * (1.0f / 13.0f));
}

#if HAVE_TDM
// Issue one TDM tile load: 128 rows x 32 f16, row stride lda elems, with LDS
// padding of 4 dwords after every 16 dwords (=> 40-half LDS row stride).
__device__ __forceinline__ void tdm_load_tile(const _Float16* gptr, unsigned lds_addr,
                                              i32x8 g1) {
  uint64_t ga = (uint64_t)(uintptr_t)gptr;
  u32x4 g0;
  g0[0] = 1u;                                   // count=1, no gather/restore
  g0[1] = lds_addr;                             // LDS byte address (low32 of flat)
  g0[2] = (unsigned)ga;                         // global_addr[31:0] at bits 95:64
  g0[3] = (unsigned)((ga >> 32) & 0x01FFFFFFu)  // global_addr[56:32]
        | 0x80000000u;                          // type=2 ("image") bits 127:126
  i32x4 z4 = {0, 0, 0, 0};
#if __clang_major__ >= 23
  i32x8 z8 = {0, 0, 0, 0, 0, 0, 0, 0};
  __builtin_amdgcn_tensor_load_to_lds(g0, g1, z4, z4, z8, 0);
#else
  __builtin_amdgcn_tensor_load_to_lds(g0, g1, z4, z4, 0);
#endif
}
#endif

// ---------- generic WMMA GEMM: C = act(A[M,K] @ B[K,N] + bias) ----------
// Block: 256 threads (8 waves). Wave w owns rows [blk*128+16w, +16) and a
// 16 x (NW*16) output strip: one A fragment feeds NW WMMAs per k-step.
template <int NW, int KMAX>
__global__ __launch_bounds__(256)
void wmma_gemm(const _Float16* __restrict__ A, int lda,
               const _Float16* __restrict__ B, int ldb,
               _Float16* __restrict__ Ch, float* __restrict__ Cf, int ldc,
               int K, const float* __restrict__ bias, int do_relu) {
  __shared__ __align__(16) _Float16 Bt[NW * 16][KMAX + 8]; // B^T chunk
  __shared__ __align__(16) _Float16 As[2][128][40];        // 32 k-halves + 8 pad

  const int tid  = threadIdx.x;
  const int lane = tid & 31;
  const int wave = tid >> 5;
  const int rowbase = blockIdx.x * 128;
  const int nbase   = blockIdx.y * (NW * 16);

  // Stage B^T for this column strip (reused across whole K loop).
  for (int idx = tid; idx < NW * 16 * K; idx += 256) {
    int n = idx / K;
    int k = idx - n * K;
    Bt[n][k] = B[(size_t)k * ldb + nbase + n];
  }

  const int m  = lane & 15;   // row in wave tile / C column index
  const int hi = lane >> 4;   // half-wave select (contiguous k = 16*hi + s)

  v8f acc[NW] = {};

#if HAVE_TDM
  // ---- async A-tile staging via Tensor Data Mover, double buffered ----
  const unsigned lds0 = (unsigned)(uintptr_t)&As[0][0][0];
  const unsigned lds1 = (unsigned)(uintptr_t)&As[1][0][0];
  i32x8 g1;
  g1[0] = (1 << 16)              // data_size = 2 bytes
        | (1 << 20)              // pad_enable
        | (3 << 22)              // pad_interval: every 16 dwords (64B row)
        | (3 << 25);             // pad_amount: 4 dwords (16B) -> 80B stride
  g1[1] = (32 << 16);            // tensor_dim0 = 32 elems
  g1[2] = (128 << 16);           // tensor_dim1 = 128 rows
  g1[3] = (32 << 16);            // tile_dim0 = 32
  g1[4] = 128;                   // tile_dim1 = 128
  g1[5] = lda;                   // tensor_dim0_stride (elements)
  g1[6] = 0;
  g1[7] = 0;

  if (wave == 0)
    tdm_load_tile(A + (size_t)rowbase * lda, lds0, g1);

  for (int kb = 0; kb < K; kb += 32) {
    const int cur = (kb >> 5) & 1;
    if (wave == 0) {
      if (kb + 32 < K) {
        tdm_load_tile(A + (size_t)rowbase * lda + kb + 32, cur ? lds0 : lds1, g1);
        if (kb + 64 < K)
          __builtin_prefetch(A + (size_t)rowbase * lda + kb + 64, 0, 1);
        __builtin_amdgcn_s_wait_tensorcnt(1);  // current tile landed
      } else {
        __builtin_amdgcn_s_wait_tensorcnt(0);
      }
    }
    __syncthreads();  // release current buffer (and Bt on first pass)

    FragH a;
    const _Float16* arow = &As[cur][wave * 16 + m][0];
    a.q[0] = *(const v4u*)(arow + hi * 16);
    a.q[1] = *(const v4u*)(arow + hi * 16 + 8);
    #pragma unroll
    for (int t = 0; t < NW; ++t) {
      FragH b;
      const _Float16* brow = &Bt[t * 16 + m][kb + hi * 16];
      b.q[0] = ((const v4u*)brow)[0];
      b.q[1] = ((const v4u*)brow)[1];
      acc[t] = __builtin_amdgcn_wmma_f32_16x16x32_f16(false, a.v, false, b.v,
                                                      (short)0, acc[t], false, false);
    }
    __syncthreads();  // all reads done before buffer is re-filled
  }
#else
  // ---- fallback: manual global->reg->LDS staging (single buffer) ----
  const int srow  = tid >> 1;
  const int shalf = tid & 1;
  for (int kb = 0; kb < K; kb += 32) {
    const _Float16* ap = A + (size_t)(rowbase + srow) * lda + kb + shalf * 16;
    v4u d0 = ((const v4u*)ap)[0];
    v4u d1 = ((const v4u*)ap)[1];
    if (kb + 32 < K) __builtin_prefetch(ap + 32, 0, 1);
    __syncthreads();
    *(v4u*)&As[0][srow][shalf * 16]     = d0;
    *(v4u*)&As[0][srow][shalf * 16 + 8] = d1;
    __syncthreads();

    FragH a;
    const _Float16* arow = &As[0][wave * 16 + m][0];
    a.q[0] = *(const v4u*)(arow + hi * 16);
    a.q[1] = *(const v4u*)(arow + hi * 16 + 8);
    #pragma unroll
    for (int t = 0; t < NW; ++t) {
      FragH b;
      const _Float16* brow = &Bt[t * 16 + m][kb + hi * 16];
      b.q[0] = ((const v4u*)brow)[0];
      b.q[1] = ((const v4u*)brow)[1];
      acc[t] = __builtin_amdgcn_wmma_f32_16x16x32_f16(false, a.v, false, b.v,
                                                      (short)0, acc[t], false, false);
    }
  }
#endif

  // Epilogue: C layout — VGPR r: M = r + 8*hi, N = lane&15 (per 16x16 tile)
  #pragma unroll
  for (int t = 0; t < NW; ++t) {
    const int col = nbase + t * 16 + m;
    const float bv = bias ? bias[col] : 0.0f;
    #pragma unroll
    for (int r = 0; r < 8; ++r) {
      int row = rowbase + wave * 16 + r + hi * 8;
      float v = acc[t][r] + bv;
      if (do_relu) v = fmaxf(v, 0.0f);
      if (Ch) Ch[(size_t)row * ldc + col] = (_Float16)v;
      if (Cf) Cf[(size_t)row * ldc + col] = v;
    }
  }
}

// ---------- circulant sliding-window sum over u[*,64]; optional m1 build ----------
__global__ void window_sum(const _Float16* __restrict__ u, int dir, int width,
                           _Float16* __restrict__ m1out,      // nullable, [NPAD,128] = [s/25 | u/25]
                           _Float16* __restrict__ totalout,   // total_h + col offset (stride TOTW)
                           float inv_norm) {
  int gid = blockIdx.x * blockDim.x + threadIdx.x;
  if (gid >= NPAD * MSG) return;
  int v = gid >> 6;
  int j = gid & 63;
  int base = v % NV;
  float s = 0.0f;
  for (int w = 0; w < width; ++w) {
    int r = base + dir * w;
    r += (r < 0) ? NV : 0;
    r -= (r >= NV) ? NV : 0;
    s += (float)u[(size_t)r * MSG + j];
  }
  totalout[(size_t)v * TOTW + j] = (_Float16)s;
  if (m1out) {
    m1out[(size_t)v * 128 + j]      = (_Float16)(s * inv_norm);
    m1out[(size_t)v * 128 + 64 + j] = (_Float16)((float)u[(size_t)base * MSG + j] * inv_norm);
  }
}

// ---------- log_softmax(row of 16) * mask ----------
__global__ void logsoftmax_mask_k(const float* __restrict__ logits,
                                  const unsigned char* __restrict__ mask,
                                  float* __restrict__ out) {
  int v = blockIdx.x * blockDim.x + threadIdx.x;
  if (v >= NV) return;
  const float* x = logits + (size_t)v * NCLS;
  float mx = x[0];
  #pragma unroll
  for (int i = 1; i < NCLS; ++i) mx = fmaxf(mx, x[i]);
  float se = 0.0f;
  #pragma unroll
  for (int i = 0; i < NCLS; ++i) se += __expf(x[i] - mx);
  float lse = mx + __logf(se);
  float mk = mask[v] ? 1.0f : 0.0f;
  #pragma unroll
  for (int i = 0; i < NCLS; ++i) out[(size_t)v * NCLS + i] = (x[i] - lse) * mk;
}

extern "C" void kernel_launch(void* const* d_in, const int* in_sizes, int n_in,
                              void* d_out, int out_size, void* d_ws, size_t ws_size,
                              hipStream_t stream) {
  (void)in_sizes; (void)n_in; (void)out_size; (void)ws_size;
  const float* sparse = (const float*)d_in[0];
  const float* fc1_w  = (const float*)d_in[1];
  const float* fc1_b  = (const float*)d_in[2];
  const float* mw0a   = (const float*)d_in[3];
  const float* mw0b   = (const float*)d_in[4];
  const float* mw1a   = (const float*)d_in[5];
  const float* mw1b   = (const float*)d_in[6];
  const float* rw0a   = (const float*)d_in[7];
  const float* rw0b   = (const float*)d_in[8];
  const float* rw1a   = (const float*)d_in[9];
  const float* rw1b   = (const float*)d_in[10];
  const float* fc2_w  = (const float*)d_in[11];
  const float* fc2_b  = (const float*)d_in[12];
  const unsigned char* mask = (const unsigned char*)d_in[20];

  // ---- workspace carve (~27 MB) ----
  char* ws = (char*)d_ws;
  size_t off = 0;
  auto carve = [&](size_t bytes) {
    char* p = ws + off;
    off = (off + bytes + 255) & ~(size_t)255;
    return p;
  };
  _Float16* Xh   = (_Float16*)carve((size_t)NPAD * C_IN * 2);
  _Float16* Th   = (_Float16*)carve((size_t)NPAD * TOTW * 2);  // [dense|s0f|s1f|s0r|s1r]
  _Float16* t_h  = (_Float16*)carve((size_t)NPAD * 128 * 2);
  _Float16* u0h  = (_Float16*)carve((size_t)NPAD * 64 * 2);
  _Float16* u1h  = (_Float16*)carve((size_t)NPAD * 64 * 2);
  _Float16* m1h  = (_Float16*)carve((size_t)NPAD * 128 * 2);
  float*    lg   = (float*)   carve((size_t)NPAD * NCLS * 4);
  _Float16* fc1h = (_Float16*)carve(512 * 128 * 2);
  _Float16* w0aF = (_Float16*)carve(128 * 128 * 2);
  _Float16* w0bF = (_Float16*)carve(128 * 64 * 2);
  _Float16* w1aF = (_Float16*)carve(128 * 128 * 2);
  _Float16* w1bF = (_Float16*)carve(128 * 64 * 2);
  _Float16* w0aR = (_Float16*)carve(128 * 128 * 2);
  _Float16* w0bR = (_Float16*)carve(128 * 64 * 2);
  _Float16* w1aR = (_Float16*)carve(128 * 128 * 2);
  _Float16* w1bR = (_Float16*)carve(128 * 64 * 2);
  _Float16* fc2h = (_Float16*)carve(TOTW * NCLS * 2);

  auto cvt = [&](const float* s, _Float16* d, int rv, int rt, int c) {
    int tot = rt * c;
    cvt_f32_f16_pad<<<(tot + 255) / 256, 256, 0, stream>>>(s, d, rv, rt, c);
  };
  cvt(sparse, Xh, NV, NPAD, C_IN);
  cvt(fc1_w, fc1h, 512, 512, 128);
  make_w0a_eff<<<64, 256, 0, stream>>>(mw0a, w0aF);
  make_w0a_eff<<<64, 256, 0, stream>>>(rw0a, w0aR);
  cvt(mw0b, w0bF, 128, 128, 64);
  cvt(mw1a, w1aF, 128, 128, 128);
  cvt(mw1b, w1bF, 128, 128, 64);
  cvt(rw0b, w0bR, 128, 128, 64);
  cvt(rw1a, w1aR, 128, 128, 128);
  cvt(rw1b, w1bR, 128, 128, 64);
  cvt(fc2_w, fc2h, TOTW, TOTW, NCLS);

  // NW=4 strip GEMM (Nout multiple of 64), two KMAX variants to bound LDS.
  auto gemm512 = [&](const _Float16* A, int lda, const _Float16* B, int ldb,
                     _Float16* Ch, float* Cf, int ldc, int K, int Nout,
                     const float* bias, int relu) {
    dim3 grid(NPAD / 128, Nout / 64);
    wmma_gemm<4, 512><<<grid, 256, 0, stream>>>(A, lda, B, ldb, Ch, Cf, ldc, K, bias, relu);
  };
  auto gemm128 = [&](const _Float16* A, int lda, const _Float16* B, int ldb,
                     _Float16* Ch, float* Cf, int ldc, int K, int Nout,
                     const float* bias, int relu) {
    dim3 grid(NPAD / 128, Nout / 64);
    wmma_gemm<4, 128><<<grid, 256, 0, stream>>>(A, lda, B, ldb, Ch, Cf, ldc, K, bias, relu);
  };

  // dense = relu(X @ fc1_w + b), written into total[:, 0:128]
  gemm512(Xh, C_IN, fc1h, 128, Th, nullptr, TOTW, C_IN, 128, fc1_b, 1);

  const int WIN_BLKS = (NPAD * MSG + 255) / 256;
  auto branch = [&](const _Float16* w0a, const _Float16* w0b,
                    const _Float16* w1a, const _Float16* w1b,
                    int dir, int colS0, int colS1) {
    gemm128(Th, TOTW, w0a, 128, t_h, nullptr, 128, 128, 128, nullptr, 1);  // t0
    gemm128(t_h, 128, w0b, 64, u0h, nullptr, 64, 128, 64, nullptr, 1);     // u0
    window_sum<<<WIN_BLKS, 256, 0, stream>>>(u0h, dir, 13, m1h, Th + colS0, 1.0f / 25.0f);
    gemm128(m1h, 128, w1a, 128, t_h, nullptr, 128, 128, 128, nullptr, 1);  // t1
    gemm128(t_h, 128, w1b, 64, u1h, nullptr, 64, 128, 64, nullptr, 1);     // u1
    window_sum<<<WIN_BLKS, 256, 0, stream>>>(u1h, dir, 25, nullptr, Th + colS1, 0.0f);
  };
  branch(w0aF, w0bF, w1aF, w1bF, +1, 128, 192);   // forward branch
  branch(w0aR, w0bR, w1aR, w1bR, -1, 256, 320);   // reverse branch

  // logits = total @ fc2_w + fc2_b  (f32 out, N=16 -> NW=1 instantiation)
  {
    dim3 grid(NPAD / 128, 1);
    wmma_gemm<1, 384><<<grid, 256, 0, stream>>>(Th, TOTW, fc2h, NCLS, nullptr, lg,
                                                NCLS, TOTW, fc2_b, 0);
  }
  logsoftmax_mask_k<<<(NV + 255) / 256, 256, 0, stream>>>(lg, mask, (float*)d_out);
}
